// CrossAttention_28329604284446
// MI455X (gfx1250) — compile-verified, matrix operand order checked by reading, not run on hardware
//
#include <hip/hip_runtime.h>
#include <hip/hip_bf16.h>

// ---------------------------------------------------------------------------
// CrossAttention + structure bias for MI455X (gfx1250, wave32, WMMA).
//   B=4, L=1024, HID=1024, H=16, DH=64, BH=64.
// Pipeline:
//   1) proj_gemm  x3 : Q = X@Wq^T+bq (bf16), K (bf16), V (bf16, stored [bh,d,r])
//   2) scores_kernel : per (l-tile, r-tile): LDS-stage S tile once, loop 64 bh:
//                      QK^T via v_wmma_f32_16x16x32_bf16, + VALU bias terms,
//                      scale + mask, write scores (second output).
//   3) softmax_pv    : per (bh, l-tile): softmax over r, probs->LDS bf16,
//                      ctx = P@V via WMMA; V^T chunks double-buffered through
//                      LDS with global_load_async_to_lds_b128 / s_wait_asynccnt.
//   4) proj_gemm     : out = ctx@Wd^T + bd (f32 -> d_out)
//   5) ln_kernel     : residual + LayerNorm in place.
// Workspace: Qbf 8MB | Kbf 8MB | Vtbf 8MB | ctx(f32) 16MB  = 40MB.
// ---------------------------------------------------------------------------

typedef __bf16 bf16_t;
typedef bf16_t bf16x16 __attribute__((ext_vector_type(16)));
typedef bf16_t bf16x8  __attribute__((ext_vector_type(8)));
typedef float  f32x8   __attribute__((ext_vector_type(8)));
typedef float  f32x4   __attribute__((ext_vector_type(4)));

#define CA_B   4
#define CA_L   1024
#define CA_HID 1024
#define CA_H   16
#define CA_DH  64
#define CA_BH  (CA_B * CA_H)
#define CA_SCALE 0.125f   // 1/sqrt(64)

// ---------------------------------------------------------------------------
// WMMA fragment helpers.
// CDNA5 16x16x32 bf16 A layout (per ISA 7.12.2): lane = g*16 + m (g=lane>>4),
// lane's 16 elements cover k = {k0+g*8 .. +7} and {k0+16+g*8 .. +7}:
// two contiguous runs of 8. B mirrors with n = lane&15, contiguous in k.
// C/D: vgpr i -> row m = i + 8*g, col n = lane&15.
// ---------------------------------------------------------------------------
__device__ __forceinline__ bf16x16 frag_from_f32(const float* __restrict__ mat,
                                                 int row0, int rstride, int k0) {
    int lane = threadIdx.x & 31;
    const float* rp = mat + (size_t)(row0 + (lane & 15)) * rstride + k0 + ((lane >> 4) << 3);
    f32x4 a0 = *(const f32x4*)(rp);
    f32x4 a1 = *(const f32x4*)(rp + 4);
    f32x4 b0 = *(const f32x4*)(rp + 16);
    f32x4 b1 = *(const f32x4*)(rp + 20);
    bf16x16 f;
#pragma unroll
    for (int i = 0; i < 4; ++i) {
        f[i]      = (bf16_t)a0[i];
        f[4 + i]  = (bf16_t)a1[i];
        f[8 + i]  = (bf16_t)b0[i];
        f[12 + i] = (bf16_t)b1[i];
    }
    return f;
}

__device__ __forceinline__ bf16x16 frag_from_bf16(const bf16_t* mat,
                                                  int row0, int rstride, int k0) {
    int lane = threadIdx.x & 31;
    const bf16_t* rp = mat + (size_t)(row0 + (lane & 15)) * rstride + k0 + ((lane >> 4) << 3);
    bf16x8 lo = *(const bf16x8*)(rp);
    bf16x8 hi = *(const bf16x8*)(rp + 16);
    bf16x16 f;
#pragma unroll
    for (int i = 0; i < 8; ++i) { f[i] = lo[i]; f[8 + i] = hi[i]; }
    return f;
}

__device__ __forceinline__ f32x8 wmma_bf16(bf16x16 a, bf16x16 b, f32x8 c) {
    return __builtin_amdgcn_wmma_f32_16x16x32_bf16(false, a, false, b, (short)0, c,
                                                   false, false);
}

// Async global->LDS copy (gfx1250 VGLOBAL async op, tracked by ASYNCcnt).
__device__ __forceinline__ void async_ld_b128(unsigned lds_addr, const void* gptr) {
    asm volatile("global_load_async_to_lds_b128 %0, %1, off"
                 :: "v"(lds_addr), "v"(gptr) : "memory");
}

// ---------------------------------------------------------------------------
// Kernel 1/4: out[m,n] = sum_k X[m,k]*W[n,k] + bias[n].  M=4096,N=1024,K=1024.
// 256 threads = 8 waves; wave w owns a 16x32 strip (two 16x16 C tiles).
// Grid: (M/16, N/256).  mode: 0 = bf16 [m,n]; 1 = bf16 V^T [bh,d,l]; 2 = f32.
// ---------------------------------------------------------------------------
__global__ __launch_bounds__(256) void proj_gemm(const float* __restrict__ X,
                                                 const float* __restrict__ W,
                                                 const float* __restrict__ bias,
                                                 void* out, int mode) {
    int wv   = threadIdx.x >> 5;
    int m0   = blockIdx.x * 16;
    int nb0  = blockIdx.y * 256 + wv * 32;

    f32x8 c0 = {}, c1 = {};
#pragma unroll 4
    for (int k0 = 0; k0 < CA_HID; k0 += 32) {
        bf16x16 a  = frag_from_f32(X, m0, CA_HID, k0);
        bf16x16 b0 = frag_from_f32(W, nb0, CA_HID, k0);
        bf16x16 b1 = frag_from_f32(W, nb0 + 16, CA_HID, k0);
        c0 = wmma_bf16(a, b0, c0);
        c1 = wmma_bf16(a, b1, c1);
    }

    int lane = threadIdx.x & 31;
    int g    = lane >> 4;
#pragma unroll
    for (int t = 0; t < 2; ++t) {
        f32x8 c  = (t == 0) ? c0 : c1;
        int   nb = nb0 + t * 16;
        int   n  = nb + (lane & 15);
        float bv = bias[n];
#pragma unroll
        for (int i = 0; i < 8; ++i) {
            int   m   = m0 + i + 8 * g;
            float val = c[i] + bv;
            if (mode == 0) {
                ((bf16_t*)out)[(size_t)m * 1024 + n] = (bf16_t)val;
            } else if (mode == 1) {
                int bb = m >> 10, l = m & 1023, hh = n >> 6, dd = n & 63;
                ((bf16_t*)out)[(((size_t)(bb * 16 + hh) * 64 + dd) << 10) + l] = (bf16_t)val;
            } else {
                ((float*)out)[(size_t)m * 1024 + n] = val;
            }
        }
    }
}

// ---------------------------------------------------------------------------
// Kernel 2: scores[b,h,l,r] = (q.k + q.S + k.S)/8 + mask[b,r].
// Grid (L/16, L/16): one 16x16 (l,r) tile; S tile staged to LDS as bf16 ONCE,
// then loop all 64 (b,h): 2x WMMA for QK^T, VALU dot-products for the two
// diagonal bias einsums against the LDS S tile.  S read exactly once (268MB).
// ---------------------------------------------------------------------------
__global__ __launch_bounds__(256) void scores_kernel(const bf16_t* __restrict__ Qbf,
                                                     const bf16_t* __restrict__ Kbf,
                                                     const float* __restrict__ S,
                                                     const float* __restrict__ mask,
                                                     float* __restrict__ scoresOut) {
    __shared__ __align__(16) bf16_t sS[16 * 16 * 64];  // 32 KB: S[l,r,d] tile

    int l0 = blockIdx.x * 16;
    int r0 = blockIdx.y * 16;

    // Cooperative S-tile load: thread t owns (l,r) pair, 64 contiguous floats.
    {
        int t = threadIdx.x;
        int ll = t >> 4, rr = t & 15;
        const float* srow = S + ((size_t)(l0 + ll) * CA_L + (r0 + rr)) * CA_DH;
        bf16_t* dst = &sS[(ll * 16 + rr) * 64];
#pragma unroll
        for (int d = 0; d < 64; d += 4) {
            f32x4 v = *(const f32x4*)(srow + d);
            dst[d]     = (bf16_t)v[0];
            dst[d + 1] = (bf16_t)v[1];
            dst[d + 2] = (bf16_t)v[2];
            dst[d + 3] = (bf16_t)v[3];
        }
    }
    __syncthreads();

    int wv   = threadIdx.x >> 5;
    int lane = threadIdx.x & 31;
    int g    = lane >> 4;
    int n    = lane & 15;

    for (int bh = wv; bh < CA_BH; bh += 8) {
        int b = bh >> 4, h = bh & 15;
        const bf16_t* Qm = Qbf + h * 64;   // row index = b*L + l, stride 1024
        const bf16_t* Km = Kbf + h * 64;

        // prefetch the next bh's Q rows into cache while we compute
        if (bh + 8 < CA_BH) {
            int bn = (bh + 8) >> 4, hn = (bh + 8) & 15;
            __builtin_prefetch(Qbf + ((size_t)(bn * CA_L + l0 + n) * 1024) + hn * 64, 0, 0);
        }

        bf16x16 aq0 = frag_from_bf16(Qm, b * CA_L + l0, 1024, 0);
        bf16x16 aq1 = frag_from_bf16(Qm, b * CA_L + l0, 1024, 32);
        bf16x16 bk0 = frag_from_bf16(Km, b * CA_L + r0, 1024, 0);
        bf16x16 bk1 = frag_from_bf16(Km, b * CA_L + r0, 1024, 32);

        f32x8 c = {};
        c = wmma_bf16(aq0, bk0, c);
        c = wmma_bf16(aq1, bk1, c);

        int   r    = r0 + n;
        float mv   = mask[b * CA_L + r];
        const bf16_t* krow = Kbf + ((size_t)(b * CA_L + r)) * 1024 + h * 64;

#pragma unroll
        for (int i = 0; i < 8; ++i) {
            int li = i + 8 * g;
            int l  = l0 + li;
            const bf16_t* qrow = Qbf + ((size_t)(b * CA_L + l)) * 1024 + h * 64;
            const bf16_t* srow = &sS[(li * 16 + n) * 64];
            float bq = 0.f, bk = 0.f;
#pragma unroll 8
            for (int d = 0; d < 64; ++d) {
                float s = (float)srow[d];
                bq += (float)qrow[d] * s;
                bk += (float)krow[d] * s;
            }
            float val = (c[i] + bq + bk) * CA_SCALE + mv;
            scoresOut[((size_t)bh << 20) + ((size_t)l << 10) + r] = val;
        }
    }
}

// ---------------------------------------------------------------------------
// Kernel 3: softmax over r + ctx = P @ V.  Grid (L/16, BH), 128 threads.
// Phase 1: each of 4 waves handles 4 rows: rowmax, exp, rowsum; unnormalized
//          probs staged to LDS as bf16 (A-fragment-friendly layout).
// Phase 2: wave w computes d-subtile w*16 via 32 WMMA steps; the V^T 16x32
//          bf16 chunk is double-buffered through LDS with async global->LDS
//          copies (ASYNCcnt pipeline), overlapping HBM/L2 latency with WMMA.
// ---------------------------------------------------------------------------
#define PSTR 1040   // P row stride (elements), keeps 16B alignment
#define VCH  32     // V^T r-chunk per WMMA step

__global__ __launch_bounds__(128) void softmax_pv(const float* __restrict__ scores,
                                                  const bf16_t* __restrict__ Vtbf,
                                                  float* __restrict__ ctx) {
    __shared__ __align__(16) bf16_t Pl[16][PSTR];
    __shared__ __align__(16) bf16_t Vs[4][2][16][VCH];   // [wave][buf][d][r] 8KB
    __shared__ float invsum[16];

    int l0   = blockIdx.x * 16;
    int bh   = blockIdx.y;
    int b    = bh >> 4, h = bh & 15;
    int wv   = threadIdx.x >> 5;
    int lane = threadIdx.x & 31;

    // ---- Phase 1: softmax rows ----
    for (int j = 0; j < 4; ++j) {
        int rowi = wv * 4 + j;
        const float* srow = scores + ((size_t)bh << 20) + ((size_t)(l0 + rowi) << 10);

        float mx = -3.4e38f;
        for (int cidx = lane; cidx < CA_L; cidx += 32)
            mx = fmaxf(mx, srow[cidx]);
#pragma unroll
        for (int off = 16; off > 0; off >>= 1)
            mx = fmaxf(mx, __shfl_xor(mx, off, 32));

        float sum = 0.f;
        for (int cidx = lane; cidx < CA_L; cidx += 32) {
            float e = __expf(srow[cidx] - mx);
            Pl[rowi][cidx] = (bf16_t)e;
            sum += e;
        }
#pragma unroll
        for (int off = 16; off > 0; off >>= 1)
            sum += __shfl_xor(sum, off, 32);
        if (lane == 0) invsum[rowi] = 1.f / sum;
    }
    __syncthreads();

    // ---- Phase 2: ctx[l, dsub+n] = sum_r P[l,r] * V[r,d] ----
    int dsub = wv * 16;
    const bf16_t* Vtb = Vtbf + ((size_t)bh * 64 + dsub) * 1024;   // 16 rows [d][r]

    // lane copies 16B chunks c = lane and lane+32 of the 1KB tile:
    // chunk c -> row = c>>2 (c<32: rows 0..7; else 8..15), 16B segment = c&3.
    int c0r = lane >> 2,        c0s = (lane & 3) * 8;
    int c1r = (lane + 32) >> 2, c1s = c0s;

    auto issue = [&](int r0, int buf) {
        async_ld_b128((unsigned)(size_t)&Vs[wv][buf][c0r][c0s],
                      Vtb + (size_t)c0r * 1024 + r0 + c0s);
        async_ld_b128((unsigned)(size_t)&Vs[wv][buf][c1r][c1s],
                      Vtb + (size_t)c1r * 1024 + r0 + c1s);
    };

    issue(0, 0);   // preload first chunk

    f32x8 c = {};
    for (int rr = 0; rr < CA_L; rr += VCH) {
        int buf = (rr >> 5) & 1;
        if (rr + VCH < CA_L) {
            issue(rr + VCH, buf ^ 1);
            asm volatile("s_wait_asynccnt 2" ::: "memory");  // current chunk done
        } else {
            asm volatile("s_wait_asynccnt 0" ::: "memory");
        }
        bf16x16 a = frag_from_bf16(&Pl[0][0], 0, PSTR, rr);
        bf16x16 v = frag_from_bf16(&Vs[wv][buf][0][0], 0, VCH, 0);
        c = wmma_bf16(a, v, c);
    }

    int g = lane >> 4;
    int d = dsub + (lane & 15);
#pragma unroll
    for (int i = 0; i < 8; ++i) {
        int li = i + 8 * g;
        int l  = l0 + li;
        ctx[((size_t)(b * CA_L + l)) * 1024 + h * 64 + d] = c[i] * invsum[li];
    }
}

// ---------------------------------------------------------------------------
// Kernel 5: residual + LayerNorm in place over d_out rows. Grid 4096 x 256.
// ---------------------------------------------------------------------------
__global__ __launch_bounds__(256) void ln_kernel(float* __restrict__ out,
                                                 const float* __restrict__ resid,
                                                 const float* __restrict__ gamma,
                                                 const float* __restrict__ beta) {
    __shared__ float red0[8], red1[8];
    int row = blockIdx.x;
    float*       o = out + (size_t)row * CA_HID;
    const float* r = resid + (size_t)row * CA_HID;

    float xv[4], s1 = 0.f, s2 = 0.f;
#pragma unroll
    for (int i = 0; i < 4; ++i) {
        int cidx = threadIdx.x + i * 256;
        float v = o[cidx] + r[cidx];
        xv[i] = v; s1 += v; s2 += v * v;
    }
#pragma unroll
    for (int off = 16; off > 0; off >>= 1) {
        s1 += __shfl_xor(s1, off, 32);
        s2 += __shfl_xor(s2, off, 32);
    }
    int wv = threadIdx.x >> 5, lane = threadIdx.x & 31;
    if (lane == 0) { red0[wv] = s1; red1[wv] = s2; }
    __syncthreads();
    if (threadIdx.x == 0) {
        float a = 0.f, bq = 0.f;
#pragma unroll
        for (int j = 0; j < 8; ++j) { a += red0[j]; bq += red1[j]; }
        red0[0] = a; red1[0] = bq;
    }
    __syncthreads();
    float mean = red0[0] * (1.f / CA_HID);
    float var  = red1[0] * (1.f / CA_HID) - mean * mean;
    float rs   = rsqrtf(var + 1e-5f);
#pragma unroll
    for (int i = 0; i < 4; ++i) {
        int cidx = threadIdx.x + i * 256;
        o[cidx] = (xv[i] - mean) * rs * gamma[cidx] + beta[cidx];
    }
}

// ---------------------------------------------------------------------------
extern "C" void kernel_launch(void* const* d_in, const int* in_sizes, int n_in,
                              void* d_out, int out_size, void* d_ws, size_t ws_size,
                              hipStream_t stream) {
    (void)in_sizes; (void)n_in; (void)out_size; (void)ws_size;

    const float* hsq   = (const float*)d_in[0];
    const float* hsv   = (const float*)d_in[1];
    const float* mask  = (const float*)d_in[2];
    const float* S     = (const float*)d_in[3];
    const float* Wq    = (const float*)d_in[4];
    const float* bq    = (const float*)d_in[5];
    const float* Wk    = (const float*)d_in[6];
    const float* bk    = (const float*)d_in[7];
    const float* Wv    = (const float*)d_in[8];
    const float* bv    = (const float*)d_in[9];
    const float* Wd    = (const float*)d_in[10];
    const float* bd    = (const float*)d_in[11];
    const float* gamma = (const float*)d_in[12];
    const float* beta  = (const float*)d_in[13];

    float* outF      = (float*)d_out;                        // [B,L,HID]
    float* scoresOut = outF + (size_t)CA_B * CA_L * CA_HID;  // [B,H,L,L]

    char*   w8   = (char*)d_ws;
    bf16_t* Qbf  = (bf16_t*)(w8);
    bf16_t* Kbf  = (bf16_t*)(w8 + (size_t)8  * 1024 * 1024);
    bf16_t* Vtbf = (bf16_t*)(w8 + (size_t)16 * 1024 * 1024);
    float*  ctx  = (float*)(w8 + (size_t)24 * 1024 * 1024);

    dim3 gemmGrid(256, 4);   // M/16 x N/256

    proj_gemm<<<gemmGrid, 256, 0, stream>>>(hsq, Wq, bq, (void*)Qbf, 0);
    proj_gemm<<<gemmGrid, 256, 0, stream>>>(hsv, Wk, bk, (void*)Kbf, 0);
    proj_gemm<<<gemmGrid, 256, 0, stream>>>(hsv, Wv, bv, (void*)Vtbf, 1);

    scores_kernel<<<dim3(64, 64), 256, 0, stream>>>(Qbf, Kbf, S, mask, scoresOut);

    softmax_pv<<<dim3(64, 64), 128, 0, stream>>>(scoresOut, Vtbf, ctx);

    proj_gemm<<<gemmGrid, 256, 0, stream>>>(ctx, Wd, bd, (void*)outF, 2);

    ln_kernel<<<4096, 256, 0, stream>>>(outF, hsq, gamma, beta);
}
